// GNTMoE_64768106824186
// MI455X (gfx1250) — matrix operand order repaired and checked by reading, not compiled, not run on hardware
//
#include <hip/hip_runtime.h>
#include <hip/hip_bf16.h>
#include <math.h>

// ---- problem constants (from reference) ----
#define LAY 2
#define NB  4
#define SEQ 1024
#define DM  512
#define NH  8
#define HD  64
#define NE  8
#define DHID 2048
#define TKN (NB*SEQ)          // 4096 tokens

typedef __attribute__((ext_vector_type(16))) __bf16 v16bf;
typedef __attribute__((ext_vector_type(8)))  float  v8f;
typedef __attribute__((ext_vector_type(8)))  __bf16 bf8;

// explicit global-address-space access (avoid flat_* codegen)
typedef __attribute__((address_space(1))) const bf8*    gcp8;
typedef __attribute__((address_space(1))) const float*  gcpf;
typedef __attribute__((address_space(1))) float*        gpf;
typedef __attribute__((address_space(1))) __bf16*       gph;

static __device__ inline bf8 gload8(const __bf16* p) { return *(gcp8)p; }

// optional CDNA5 async global->LDS path (compile-time guarded)
#if defined(__has_builtin)
# if __has_builtin(__builtin_amdgcn_global_load_async_to_lds_b32) && \
     __has_builtin(__builtin_amdgcn_s_wait_asynccnt)
#  define HAVE_ASYNC_LDS 1
# endif
#endif
#ifndef HAVE_ASYNC_LDS
# define HAVE_ASYNC_LDS 0
#endif
#if HAVE_ASYNC_LDS
typedef __attribute__((address_space(1))) int* gip;   // global int*
typedef __attribute__((address_space(3))) int* lip;   // LDS int*
#endif

static __device__ inline v8f wmma_bf16(v16bf a, v16bf b, v8f c) {
    // 8 args: (neg_a, A, neg_b, B, c_mod, C, reuse_a, reuse_b)
    return __builtin_amdgcn_wmma_f32_16x16x32_bf16(false, a, false, b, (short)0, c,
                                                   false, false);
}
static __device__ inline v16bf cat8(bf8 lo, bf8 hi8) {
    return __builtin_shufflevector(lo, hi8, 0,1,2,3,4,5,6,7,8,9,10,11,12,13,14,15);
}

// ---------------------------------------------------------------------------
// Generic bf16 GEMM with TRANSPOSED weights:
//   C[M,N](f32) = A[M,K] * B[K,N],  B given as BT[N,K] row-major (ldbt = K).
// A-operand (16x32): lane m holds K {hi*16 .. +15}            -> 2x b128 loads.
// B-operand (32x16): lane n holds BT[n][k0+hi*16 .. +15]     -> 2x b128 loads.
// Software-pipelined (double-buffered) k-loop; all accesses via AS(1).
// Block 256 (8 waves); tile 128(M) x 64(N); K step 32.
// ---------------------------------------------------------------------------
__global__ __launch_bounds__(256)
void k_gemm_bf16(const __bf16* __restrict__ A, int lda,
                 const __bf16* __restrict__ BT, int ldbt,
                 int M, int N, int K,
                 const float* __restrict__ bias,
                 float constScale, int doGelu,
                 const float* __restrict__ rowScale, int rsStride,
                 float* __restrict__ outF, int accumulate,
                 __bf16* __restrict__ outB)
{
    const int lane = threadIdx.x & 31;
    const int wave = threadIdx.x >> 5;
    const int m0 = blockIdx.x * 128 + wave * 16;
    const int n0 = blockIdx.y * 64;
    const int n = lane & 15, hi = lane >> 4;

    const __bf16* ap  = A  + (long)(m0 + n) * lda + hi * 16;
    const __bf16* bp0 = BT + (long)(n0 +  0 + n) * ldbt + hi * 16;
    const __bf16* bp1 = BT + (long)(n0 + 16 + n) * ldbt + hi * 16;
    const __bf16* bp2 = BT + (long)(n0 + 32 + n) * ldbt + hi * 16;
    const __bf16* bp3 = BT + (long)(n0 + 48 + n) * ldbt + hi * 16;

    v8f acc[4] = {};

    // prologue: first operand set
    v16bf a  = cat8(gload8(ap),      gload8(ap + 8));
    v16bf b0 = cat8(gload8(bp0),     gload8(bp0 + 8));
    v16bf b1 = cat8(gload8(bp1),     gload8(bp1 + 8));
    v16bf b2 = cat8(gload8(bp2),     gload8(bp2 + 8));
    v16bf b3 = cat8(gload8(bp3),     gload8(bp3 + 8));

    for (int k0 = 0; k0 < K - 32; k0 += 32) {
        ap += 32; bp0 += 32; bp1 += 32; bp2 += 32; bp3 += 32;
        // issue next iteration's loads BEFORE consuming current operands
        v16bf an  = cat8(gload8(ap),  gload8(ap + 8));
        v16bf bn0 = cat8(gload8(bp0), gload8(bp0 + 8));
        v16bf bn1 = cat8(gload8(bp1), gload8(bp1 + 8));
        v16bf bn2 = cat8(gload8(bp2), gload8(bp2 + 8));
        v16bf bn3 = cat8(gload8(bp3), gload8(bp3 + 8));
        __builtin_prefetch(ap + 64, 0, 1);
        acc[0] = wmma_bf16(a, b0, acc[0]);
        acc[1] = wmma_bf16(a, b1, acc[1]);
        acc[2] = wmma_bf16(a, b2, acc[2]);
        acc[3] = wmma_bf16(a, b3, acc[3]);
        a = an; b0 = bn0; b1 = bn1; b2 = bn2; b3 = bn3;
    }
    acc[0] = wmma_bf16(a, b0, acc[0]);
    acc[1] = wmma_bf16(a, b1, acc[1]);
    acc[2] = wmma_bf16(a, b2, acc[2]);
    acc[3] = wmma_bf16(a, b3, acc[3]);

    // epilogue (C layout: vgpr r -> row m0+r (lanes 0-15) / m0+8+r (lanes 16-31))
#pragma unroll
    for (int nt = 0; nt < 4; ++nt) {
#pragma unroll
        for (int r = 0; r < 8; ++r) {
            int row = m0 + r + 8 * hi;
            int col = n0 + nt * 16 + n;
            float v = acc[nt][r] * constScale;
            if (bias) v += *(gcpf)(bias + col);
            if (doGelu) v = 0.5f * v * (1.0f + erff(v * 0.70710678118654752f));
            float s = rowScale ? *(gcpf)(rowScale + (long)row * rsStride) : 1.0f;
            if (outF) {
                long idx = (long)row * N + col;
                float prev = accumulate ? *(gpf)(outF + idx) : 0.0f;
                *(gpf)(outF + idx) = prev + s * v;
            }
            if (outB) *(gph)(outB + (long)row * N + col) = (__bf16)v;
        }
    }
}

// ---------------------------------------------------------------------------
// Transpose-convert: out[z][n*K+k] = (bf16) in[z][k*N+n], 32x32 LDS tiles,
// coalesced on both sides. Uses async global->LDS loads when available.
// ---------------------------------------------------------------------------
__global__ __launch_bounds__(256)
void k_cvt_transpose(const float* __restrict__ in, __bf16* __restrict__ out,
                     int K, int N)
{
    __shared__ float tile[32][33];
    const long mat = blockIdx.z;
    const float* src = in + mat * (long)K * N;
    __bf16* dst = out + mat * (long)K * N;
    const int kb = blockIdx.x * 32, nb = blockIdx.y * 32;
    const int tx = threadIdx.x & 31, ty = threadIdx.x >> 5;
#if HAVE_ASYNC_LDS
#pragma unroll
    for (int r0 = 0; r0 < 32; r0 += 8) {
        float* g = const_cast<float*>(src + (long)(kb + r0 + ty) * N + nb + tx);
        int* l = (int*)&tile[r0 + ty][tx];
        __builtin_amdgcn_global_load_async_to_lds_b32((gip)g, (lip)l, 0, 0);
    }
    __builtin_amdgcn_s_wait_asynccnt(0);
    __syncthreads();
#else
#pragma unroll
    for (int r0 = 0; r0 < 32; r0 += 8)
        tile[r0 + ty][tx] = src[(long)(kb + r0 + ty) * N + nb + tx];
    __syncthreads();
#endif
#pragma unroll
    for (int r0 = 0; r0 < 32; r0 += 8)
        *(gph)(dst + (long)(nb + r0 + ty) * K + kb + tx) = (__bf16)tile[tx][r0 + ty];
}

// ---------------------------------------------------------------------------
// LayerNorm: one block (256 thr) per token row of 512; emits bf16.
// ---------------------------------------------------------------------------
__global__ __launch_bounds__(256)
void k_layernorm(const float* __restrict__ x, const float* __restrict__ w,
                 const float* __restrict__ b, __bf16* __restrict__ out)
{
    __shared__ float red[256];
    const int t = blockIdx.x;
    const float* xr = x + (long)t * DM;
    float a0 = xr[threadIdx.x], a1 = xr[threadIdx.x + 256];
    red[threadIdx.x] = a0 + a1;
    __syncthreads();
    for (int o = 128; o > 0; o >>= 1) {
        if (threadIdx.x < o) red[threadIdx.x] += red[threadIdx.x + o];
        __syncthreads();
    }
    float mu = red[0] * (1.0f / DM);
    __syncthreads();
    float d0 = a0 - mu, d1 = a1 - mu;
    red[threadIdx.x] = d0 * d0 + d1 * d1;
    __syncthreads();
    for (int o = 128; o > 0; o >>= 1) {
        if (threadIdx.x < o) red[threadIdx.x] += red[threadIdx.x + o];
        __syncthreads();
    }
    float rstd = rsqrtf(red[0] * (1.0f / DM) + 1e-6f);
    out[(long)t * DM + threadIdx.x] =
        (__bf16)(d0 * rstd * w[threadIdx.x] + b[threadIdx.x]);
    out[(long)t * DM + threadIdx.x + 256] =
        (__bf16)(d1 * rstd * w[threadIdx.x + 256] + b[threadIdx.x + 256]);
}

// ---------------------------------------------------------------------------
// Gate: logits = h @ gate_w (D x E), top-2, softmax over top-2 -> dense
// combine[T,E] (zeros except the 2 picked experts). 1 wave / token.
// ---------------------------------------------------------------------------
__global__ __launch_bounds__(256)
void k_gate(const __bf16* __restrict__ h, const __bf16* __restrict__ gw,
            float* __restrict__ combine)
{
    const int lane = threadIdx.x & 31;
    const int wave = threadIdx.x >> 5;
    const int t = blockIdx.x * 8 + wave;
    float p[NE];
#pragma unroll
    for (int e = 0; e < NE; ++e) p[e] = 0.0f;
    for (int d = lane; d < DM; d += 32) {
        float hv = (float)h[(long)t * DM + d];
#pragma unroll
        for (int e = 0; e < NE; ++e) p[e] += hv * (float)gw[d * NE + e];
    }
#pragma unroll
    for (int e = 0; e < NE; ++e)
        for (int o = 16; o > 0; o >>= 1) p[e] += __shfl_xor(p[e], o, 32);
    if (lane == 0) {
        int i0 = 0;
        for (int e = 1; e < NE; ++e) if (p[e] > p[i0]) i0 = e;
        int i1 = (i0 == 0) ? 1 : 0;
        for (int e = 0; e < NE; ++e) if (e != i0 && p[e] > p[i1]) i1 = e;
        float m = p[i0];
        float e0 = expf(p[i0] - m), e1 = expf(p[i1] - m);
        float inv = 1.0f / (e0 + e1);
        for (int e = 0; e < NE; ++e) combine[(long)t * NE + e] = 0.0f;
        combine[(long)t * NE + i0] = e0 * inv;
        combine[(long)t * NE + i1] = e1 * inv;
    }
}

// ---------------------------------------------------------------------------
// Transpose V: v[b][s][h*64+d] -> vt[b][h][d][s]  (contiguous PV B-operand)
// ---------------------------------------------------------------------------
__global__ void k_transpose_v(const __bf16* __restrict__ v, __bf16* __restrict__ vt)
{
    long i = (long)blockIdx.x * blockDim.x + threadIdx.x;
    if (i >= (long)NB * NH * HD * SEQ) return;
    int s = (int)(i % SEQ); long r = i / SEQ;
    int d = (int)(r % HD);  r /= HD;
    int h = (int)(r % NH);  int b = (int)(r / NH);
    vt[i] = v[((long)(b * SEQ + s)) * DM + h * HD + d];
}

// ---------------------------------------------------------------------------
// Attention: 1 wave / (b, head, 16-query tile). Two-pass softmax, 16x1024
// bf16 score tile in LDS (32KB << 320KB/WGP).
// ---------------------------------------------------------------------------
__global__ __launch_bounds__(32)
void k_attention(const __bf16* __restrict__ q, const __bf16* __restrict__ k,
                 const __bf16* __restrict__ vt, __bf16* __restrict__ attn)
{
    __shared__ __align__(16) __bf16 sc[16 * SEQ];   // 32KB score tile
    __shared__ float rmax[16];
    __shared__ float lrow[16];

    const int lane = threadIdx.x;
    const int qt = blockIdx.x & 63;
    const int h  = (blockIdx.x >> 6) & 7;
    const int b  = blockIdx.x >> 9;
    const int q0 = qt * 16;
    const int m = lane & 15, hi = lane >> 4;

    // Q as two A operands (d 0..31 and 32..63)
    const __bf16* qbase = q + ((long)(b * SEQ + q0 + m)) * DM + h * HD;
    v16bf a0 = cat8(gload8(qbase + hi * 16),      gload8(qbase + hi * 16 + 8));
    v16bf a1 = cat8(gload8(qbase + 32 + hi * 16), gload8(qbase + 32 + hi * 16 + 8));

    float rm[8];
#pragma unroll
    for (int r = 0; r < 8; ++r) rm[r] = -1e30f;

    // ---- pass A: S = (Q K^T)/8 ----
    for (int kt = 0; kt < SEQ / 16; ++kt) {
        const __bf16* kb = k + ((long)(b * SEQ + kt * 16 + m)) * DM + h * HD;
        v16bf b0 = cat8(gload8(kb + hi * 16),      gload8(kb + hi * 16 + 8));
        v16bf b1 = cat8(gload8(kb + 32 + hi * 16), gload8(kb + 32 + hi * 16 + 8));
        v8f s = {};
        s = wmma_bf16(a0, b0, s);
        s = wmma_bf16(a1, b1, s);
#pragma unroll
        for (int r = 0; r < 8; ++r) {
            float v = s[r] * 0.125f;               // 1/sqrt(64)
            sc[(r + 8 * hi) * SEQ + kt * 16 + m] = (__bf16)v;
            float vm = v;
            for (int o = 1; o < 16; o <<= 1) vm = fmaxf(vm, __shfl_xor(vm, o, 32));
            rm[r] = fmaxf(rm[r], vm);
        }
    }
    if (m == 0) {   // lanes 0 and 16 hold the reduced maxima of rows r / r+8
#pragma unroll
        for (int r = 0; r < 8; ++r) rmax[r + 8 * hi] = rm[r];
    }
    __syncthreads();

    // ---- pass B: P = exp(S-m), O = P V ----
    const float mrow = rmax[m];     // A-layout: lanes m and m+16 serve row m
    float lacc = 0.0f;
    v8f o4[4] = {};
    for (int kt2 = 0; kt2 < SEQ / 32; ++kt2) {
        const __bf16* srow = sc + m * SEQ + kt2 * 32;
        bf8 s0 = *(const bf8*)(srow + hi * 16);         // K = hi*16 .. +7 (LDS)
        bf8 s1 = *(const bf8*)(srow + hi * 16 + 8);     // K = hi*16+8 .. +7 (LDS)
        v16bf p;
#pragma unroll
        for (int j = 0; j < 8; ++j) {
            float e0 = expf((float)s0[j] - mrow);
            float e1 = expf((float)s1[j] - mrow);
            lacc += e0 + e1;
            p[j] = (__bf16)e0;
            p[8 + j] = (__bf16)e1;
        }
#pragma unroll
        for (int nt = 0; nt < 4; ++nt) {
            const __bf16* vb = vt + (((long)(b * NH + h) * HD + nt * 16 + m)) * SEQ
                               + kt2 * 32 + hi * 16;
            v16bf bv = cat8(gload8(vb), gload8(vb + 8));
            o4[nt] = wmma_bf16(p, bv, o4[nt]);
        }
    }
    lacc += __shfl_xor(lacc, 16, 32);
    if (hi == 0) lrow[m] = lacc;
    __syncthreads();

#pragma unroll
    for (int nt = 0; nt < 4; ++nt) {
#pragma unroll
        for (int r = 0; r < 8; ++r) {
            int row = r + 8 * hi;
            float v = o4[nt][r] / lrow[row];
            *(gph)(attn + ((long)(b * SEQ + q0 + row)) * DM + h * HD + nt * 16 + m)
                = (__bf16)v;
        }
    }
}

// ---- small utility kernels ----
__global__ void k_f32_to_bf16(const float* __restrict__ in, __bf16* __restrict__ out, long n)
{
    long i = (long)blockIdx.x * blockDim.x + threadIdx.x;
    if (i < n) out[i] = (__bf16)in[i];
}
__global__ void k_copy_f32(const float* __restrict__ in, float* __restrict__ out, long n)
{
    long i = (long)blockIdx.x * blockDim.x + threadIdx.x;
    if (i < n) out[i] = in[i];
}

// ---------------------------------------------------------------------------
extern "C" void kernel_launch(void* const* d_in, const int* in_sizes, int n_in,
                              void* d_out, int out_size, void* d_ws, size_t ws_size,
                              hipStream_t stream)
{
    const float* x      = (const float*)d_in[0];
    const float* ln1_w  = (const float*)d_in[1];
    const float* ln1_b  = (const float*)d_in[2];
    const float* wq     = (const float*)d_in[3];
    const float* wk     = (const float*)d_in[4];
    const float* wv     = (const float*)d_in[5];
    const float* wo     = (const float*)d_in[6];
    const float* bo     = (const float*)d_in[7];
    const float* ln2_w  = (const float*)d_in[8];
    const float* ln2_b  = (const float*)d_in[9];
    const float* gate_w = (const float*)d_in[10];
    const float* w1     = (const float*)d_in[11];
    const float* b1     = (const float*)d_in[12];
    const float* w2     = (const float*)d_in[13];
    const float* b2     = (const float*)d_in[14];
    float* xout = (float*)d_out;                 // residual stream (f32)

    // workspace carve-up (256B aligned)
    char* wsp = (char*)d_ws;
    auto carve = [&](size_t bytes) -> char* {
        char* p = wsp;
        wsp += (bytes + 255) & ~((size_t)255);
        return p;
    };
    const long nProj = (long)LAY * DM * DM;
    const long nGate = (long)LAY * DM * NE;
    const long nW1   = (long)LAY * NE * DM * DHID;
    const long nW2   = nW1;
    const long nTD   = (long)TKN * DM;
    const long nMid  = (long)TKN * DHID;

    __bf16* wqT  = (__bf16*)carve(nProj * 2);   // transposed bf16 weights
    __bf16* wkT  = (__bf16*)carve(nProj * 2);
    __bf16* wvT  = (__bf16*)carve(nProj * 2);
    __bf16* woT  = (__bf16*)carve(nProj * 2);
    __bf16* gw_bf = (__bf16*)carve(nGate * 2);
    __bf16* w1T  = (__bf16*)carve(nW1 * 2);
    __bf16* w2T  = (__bf16*)carve(nW2 * 2);
    __bf16* h_bf   = (__bf16*)carve(nTD * 2);
    __bf16* q_bf   = (__bf16*)carve(nTD * 2);
    __bf16* k_bf   = (__bf16*)carve(nTD * 2);
    __bf16* v_bf   = (__bf16*)carve(nTD * 2);
    __bf16* vt_bf  = (__bf16*)carve(nTD * 2);
    __bf16* at_bf  = (__bf16*)carve(nTD * 2);
    __bf16* mid_bf = (__bf16*)carve(nMid * 2);
    float*  combine = (float*)carve((long)TKN * NE * 4);

    // residual stream init + weight convert/transpose (every call; deterministic)
    k_copy_f32<<<dim3((unsigned)((nTD + 255) / 256)), dim3(256), 0, stream>>>(x, xout, nTD);
    k_cvt_transpose<<<dim3(DM/32, DM/32, LAY), dim3(256), 0, stream>>>(wq, wqT, DM, DM);
    k_cvt_transpose<<<dim3(DM/32, DM/32, LAY), dim3(256), 0, stream>>>(wk, wkT, DM, DM);
    k_cvt_transpose<<<dim3(DM/32, DM/32, LAY), dim3(256), 0, stream>>>(wv, wvT, DM, DM);
    k_cvt_transpose<<<dim3(DM/32, DM/32, LAY), dim3(256), 0, stream>>>(wo, woT, DM, DM);
    k_cvt_transpose<<<dim3(DM/32, DHID/32, LAY*NE), dim3(256), 0, stream>>>(w1, w1T, DM, DHID);
    k_cvt_transpose<<<dim3(DHID/32, DM/32, LAY*NE), dim3(256), 0, stream>>>(w2, w2T, DHID, DM);
    k_f32_to_bf16<<<dim3((unsigned)((nGate + 255) / 256)), dim3(256), 0, stream>>>(gate_w, gw_bf, nGate);

    const dim3 gemmBlk(256);
    const dim3 gridD(TKN / 128, DM / 64);     // N=512
    const dim3 gridH(TKN / 128, DHID / 64);   // N=2048

    for (int l = 0; l < LAY; ++l) {
        const __bf16* wqT_l = wqT + (long)l * DM * DM;
        const __bf16* wkT_l = wkT + (long)l * DM * DM;
        const __bf16* wvT_l = wvT + (long)l * DM * DM;
        const __bf16* woT_l = woT + (long)l * DM * DM;

        k_layernorm<<<dim3(TKN), dim3(256), 0, stream>>>(xout, ln1_w + l * DM,
                                                         ln1_b + l * DM, h_bf);
        k_gemm_bf16<<<gridD, gemmBlk, 0, stream>>>(h_bf, DM, wqT_l, DM, TKN, DM, DM,
            nullptr, 1.0f, 0, nullptr, 0, nullptr, 0, q_bf);
        k_gemm_bf16<<<gridD, gemmBlk, 0, stream>>>(h_bf, DM, wkT_l, DM, TKN, DM, DM,
            nullptr, 1.0f, 0, nullptr, 0, nullptr, 0, k_bf);
        k_gemm_bf16<<<gridD, gemmBlk, 0, stream>>>(h_bf, DM, wvT_l, DM, TKN, DM, DM,
            nullptr, 1.0f, 0, nullptr, 0, nullptr, 0, v_bf);

        k_transpose_v<<<dim3((unsigned)(((long)NB * NH * HD * SEQ + 255) / 256)),
                        dim3(256), 0, stream>>>(v_bf, vt_bf);
        k_attention<<<dim3(NB * NH * (SEQ / 16)), dim3(32), 0, stream>>>(q_bf, k_bf,
                                                                          vt_bf, at_bf);
        k_gemm_bf16<<<gridD, gemmBlk, 0, stream>>>(at_bf, DM, woT_l, DM, TKN, DM, DM,
            bo + (long)l * DM, 1.0f, 0, nullptr, 0, xout, 1, nullptr);

        k_layernorm<<<dim3(TKN), dim3(256), 0, stream>>>(xout, ln2_w + l * DM,
                                                         ln2_b + l * DM, h_bf);
        k_gate<<<dim3(TKN / 8), dim3(256), 0, stream>>>(h_bf, gw_bf + (long)l * DM * NE,
                                                        combine);

        for (int e = 0; e < NE; ++e) {
            const __bf16* w1T_e = w1T + ((long)(l * NE + e)) * DM * DHID;
            const __bf16* w2T_e = w2T + ((long)(l * NE + e)) * DHID * DM;
            const float*  b1_e = b1 + ((long)(l * NE + e)) * DHID;
            const float*  b2_e = b2 + ((long)(l * NE + e)) * DM;
            k_gemm_bf16<<<gridH, gemmBlk, 0, stream>>>(h_bf, DM, w1T_e, DM,
                TKN, DHID, DM, b1_e, 1.0f, /*gelu=*/1, nullptr, 0, nullptr, 0, mid_bf);
            k_gemm_bf16<<<gridD, gemmBlk, 0, stream>>>(mid_bf, DHID, w2T_e, DHID,
                TKN, DM, DHID, b2_e, 1.0f, 0, /*rowScale=*/combine + e, NE,
                xout, /*accumulate=*/1, nullptr);
        }
    }
}
